// MambaClassifier_62165356642443
// MI455X (gfx1250) — matrix operand order repaired and checked by reading, not compiled, not run on hardware
//
#include <hip/hip_runtime.h>
#include <hip/hip_bf16.h>
#include <stdint.h>

// ---------------- model constants ----------------
constexpr int kB = 32, kL = 512, kS = 37;
constexpr int kD = 768, kDI = 1536, kN = 16, kConvK = 4, kDTR = 48, kNL = 4, kV = 32;
constexpr int kM = kB * kL;        // 16384 token rows
constexpr int kF = 2 * kS;         // 74 features
constexpr int kP  = kDTR + 2 * kN; // 80  logical proj width (dt_r | B | C)
constexpr int kPP = 128;           // proj width padded to a multiple of 64
constexpr int kDTRP = 64;          // dt_r K padded to a multiple of 32

// ---------------- WMMA types ----------------
typedef __attribute__((ext_vector_type(16))) __bf16 v16bf;
typedef __attribute__((ext_vector_type(8)))  float  v8f;
typedef __attribute__((ext_vector_type(4)))  unsigned int v4u;

union Frag16 { v16bf v; unsigned short h[16]; v4u q[2]; };

__device__ __forceinline__ unsigned short f2bf(float f) {
  unsigned int u = __float_as_uint(f);
  u += 0x7FFFu + ((u >> 16) & 1u);       // round-to-nearest-even
  return (unsigned short)(u >> 16);
}

// =====================================================================
// GEMM: C[m,n] (+)= sum_k A[m,k] * W[n,k]   A,W bf16 row-major, C f32.
// One wave per (16*MT)(M) x (16*NT)(N) tile: MT*NT accumulators, each B
// fragment reused MT times and each A fragment NT times (cuts L2 traffic).
// Requires K % 32 == 0, N % (16*NT) == 0, M % (16*MT) == 0 (by construction
// -> fully branchless body, EXEC all-ones at every WMMA).
// Fragment layouts per CDNA5 ISA §7.12.2:
//   A 16x32: lane<16 holds K {k0..k0+7, k0+16..k0+23}, lane>=16 offset +8.
//   B 32x16: lane serves column N=lane&15, K half selected by lane>>4.
//   C 16x16 f32: VGPR r -> row M = r + 8*(lane>>4), col N = lane&15.
// =====================================================================
template <int ACC, int NT, int MT>
__global__ __launch_bounds__(32)
void gemm_bf16_wmma(const unsigned short* __restrict__ A,
                    const unsigned short* __restrict__ Bw,
                    float* __restrict__ C,
                    int K, int lda, int ldb, int ldc)
{
  const int lane = threadIdx.x & 31;
  const int half = lane >> 4;
  const int l16  = lane & 15;
  const int tm0  = blockIdx.y * MT;
  const int tn0  = blockIdx.x * NT;

  const unsigned short* arow[MT];
#pragma unroll
  for (int i = 0; i < MT; ++i)
    arow[i] = A + (size_t)((tm0 + i) * 16 + l16) * lda;

  v8f c[MT][NT] = {};

  for (int k0 = 0; k0 < K; k0 += 32) {
    Frag16 a[MT];
#pragma unroll
    for (int i = 0; i < MT; ++i) {
      a[i].q[0] = *(const v4u*)(arow[i] + k0 + half * 8);
      a[i].q[1] = *(const v4u*)(arow[i] + k0 + 16 + half * 8);
    }
    if (k0 + 64 <= K) __builtin_prefetch(arow[0] + k0 + 32, 0, 3);
#pragma unroll
    for (int j = 0; j < NT; ++j) {
      Frag16 b;
      const unsigned short* browp =
          Bw + (size_t)((tn0 + j) * 16 + l16) * ldb + half * 8 + k0;
      b.q[0] = *(const v4u*)(browp);
      b.q[1] = *(const v4u*)(browp + 16);
#pragma unroll
      for (int i = 0; i < MT; ++i)
        c[i][j] = __builtin_amdgcn_wmma_f32_16x16x32_bf16(false, a[i].v, false, b.v,
                                                          (short)0, c[i][j], false, false);
    }
  }

#pragma unroll
  for (int i = 0; i < MT; ++i) {
#pragma unroll
    for (int j = 0; j < NT; ++j) {
      int col = (tn0 + j) * 16 + l16;
#pragma unroll
      for (int r = 0; r < 8; ++r) {
        int m = (tm0 + i) * 16 + half * 8 + r;
        size_t o = (size_t)m * ldc + col;
        float vv = c[i][j][r];
        if (ACC) C[o] += vv; else C[o] = vv;
      }
    }
  }
}

// =====================================================================
// Fused embedding + static projection + LayerNorm. One block per token.
// =====================================================================
__global__ __launch_bounds__(256)
void embed_layernorm(const float* __restrict__ x, const float* __restrict__ st,
                     const int* __restrict__ mask,
                     const float* __restrict__ emb_w, const float* __restrict__ emb_b,
                     const float* __restrict__ sw, const float* __restrict__ sb,
                     const float* __restrict__ lnw, const float* __restrict__ lnb,
                     float* __restrict__ h)
{
  const int m = blockIdx.x;           // token row in [0, B*L)
  const int b = m / kL;
  const int tid = threadIdx.x;
  __shared__ float sf[kF];
  __shared__ float ss[8];
  __shared__ float red[256];
  if (tid < kS)          sf[tid] = x[(size_t)m * kS + tid];
  else if (tid < kF)     sf[tid] = (float)mask[(size_t)m * kS + (tid - kS)];
  else if (tid < kF + 8) ss[tid - kF] = st[b * 8 + (tid - kF)];
  __syncthreads();

  float v[3];
  float psum = 0.f;
#pragma unroll
  for (int j = 0; j < 3; ++j) {
    int d = j * 256 + tid;
    float acc = emb_b[d] + sb[d];
    const float* ew = emb_w + (size_t)d * kF;
    for (int f = 0; f < kF; ++f) acc += sf[f] * ew[f];
    const float* swd = sw + (size_t)d * 8;
#pragma unroll
    for (int s2 = 0; s2 < 8; ++s2) acc += ss[s2] * swd[s2];
    v[j] = acc;
    psum += acc;
  }
  red[tid] = psum; __syncthreads();
  for (int o = 128; o > 0; o >>= 1) { if (tid < o) red[tid] += red[tid + o]; __syncthreads(); }
  float mu = red[0] / (float)kD;
  __syncthreads();
  float pss = 0.f;
#pragma unroll
  for (int j = 0; j < 3; ++j) { float t = v[j] - mu; pss += t * t; }
  red[tid] = pss; __syncthreads();
  for (int o = 128; o > 0; o >>= 1) { if (tid < o) red[tid] += red[tid + o]; __syncthreads(); }
  float rs = rsqrtf(red[0] / (float)kD + 1e-5f);
#pragma unroll
  for (int j = 0; j < 3; ++j) {
    int d = j * 256 + tid;
    h[(size_t)m * kD + d] = (v[j] - mu) * rs * lnw[d] + lnb[d];
  }
}

// RMSNorm producing bf16 GEMM operand. One block per token.
__global__ __launch_bounds__(256)
void rmsnorm_to_bf16(const float* __restrict__ h, const float* __restrict__ w,
                     unsigned short* __restrict__ out)
{
  const int m = blockIdx.x;
  const int tid = threadIdx.x;
  __shared__ float red[256];
  float v[3]; float pss = 0.f;
#pragma unroll
  for (int j = 0; j < 3; ++j) { v[j] = h[(size_t)m * kD + j * 256 + tid]; pss += v[j] * v[j]; }
  red[tid] = pss; __syncthreads();
  for (int o = 128; o > 0; o >>= 1) { if (tid < o) red[tid] += red[tid + o]; __syncthreads(); }
  float rs = rsqrtf(red[0] / (float)kD + 1e-5f);
#pragma unroll
  for (int j = 0; j < 3; ++j) {
    int d = j * 256 + tid;
    out[(size_t)m * kD + d] = f2bf(v[j] * rs * w[d]);
  }
}

__global__ __launch_bounds__(256)
void f32_to_bf16(const float* __restrict__ src, unsigned short* __restrict__ dst, size_t n)
{
  size_t i = (size_t)blockIdx.x * 256 + threadIdx.x;
  if (i < n) dst[i] = f2bf(src[i]);
}

// Column-padded bf16 conversion: src rows (stride srow_stride) of sk valid
// columns -> dst rows of dk columns, tail zero-filled.
__global__ __launch_bounds__(256)
void f32_to_bf16_padcols(const float* __restrict__ src, unsigned short* __restrict__ dst,
                         size_t rows, int sk, int dk, int srow_stride)
{
  size_t i = (size_t)blockIdx.x * 256 + threadIdx.x;
  if (i >= rows * (size_t)dk) return;
  int    col = (int)(i % dk);
  size_t row = i / dk;
  dst[i] = (col < sk) ? f2bf(src[row * (size_t)srow_stride + col]) : (unsigned short)0;
}

// Row-padded bf16 conversion: src (src_rows x cols) -> dst (dst_rows x cols),
// rows >= src_rows zero-filled.  Used to pad x_proj weights 80 -> 128 rows.
__global__ __launch_bounds__(256)
void f32_to_bf16_padtile(const float* __restrict__ src, unsigned short* __restrict__ dst,
                         size_t src_rows, size_t dst_rows, int cols)
{
  size_t i = (size_t)blockIdx.x * 256 + threadIdx.x;
  if (i >= dst_rows * (size_t)cols) return;
  size_t row = i / cols;
  dst[i] = (row < src_rows) ? f2bf(src[i]) : (unsigned short)0;
}

// Depthwise causal conv(K=4) + SiLU; reads xb = xz[..., :DI].
__global__ __launch_bounds__(256)
void conv_silu(const float* __restrict__ xz, const float* __restrict__ cw,
               const float* __restrict__ cb, float* __restrict__ xc,
               unsigned short* __restrict__ xc_bf)
{
  size_t gid = (size_t)blockIdx.x * 256 + threadIdx.x;
  if (gid >= (size_t)kM * kDI) return;
  int d = (int)(gid % kDI);
  size_t ml = gid / kDI;           // b*L + l
  int l = (int)(ml % kL);
  float acc = cb[d];
#pragma unroll
  for (int k = 0; k < kConvK; ++k) {
    int ls = l + k - (kConvK - 1);
    if (ls >= 0) {
      size_t srow = ml - (size_t)(l - ls);
      acc += cw[d * kConvK + k] * xz[srow * (size_t)(2 * kDI) + d];
    }
  }
  float s = acc / (1.f + __expf(-acc));
  xc[gid] = s;
  xc_bf[gid] = f2bf(s);
}

__global__ __launch_bounds__(256)
void dt_softplus(float* __restrict__ dt, const float* __restrict__ bias)
{
  size_t gid = (size_t)blockIdx.x * 256 + threadIdx.x;
  if (gid >= (size_t)kM * kDI) return;
  int d = (int)(gid % kDI);
  float xv = dt[gid] + bias[d];
  dt[gid] = (xv > 20.f) ? xv : log1pf(__expf(xv));
}

// Selective scan: one thread per (b,d); 16-state recurrence in registers,
// shared B_t/C_t in LDS. Fused +D*u and *silu(z); writes bf16 for out_proj.
__global__ __launch_bounds__(256)
void selective_scan(const float* __restrict__ xc, const float* __restrict__ dt,
                    const float* __restrict__ xz, const float* __restrict__ proj,
                    const float* __restrict__ Alog, const float* __restrict__ Dp,
                    unsigned short* __restrict__ y_bf)
{
  const int b = blockIdx.y;
  const int d = blockIdx.x * 256 + threadIdx.x;
  const int tid = threadIdx.x;
  __shared__ float sB[kN], sC[kN];
  float hstate[kN], a[kN];
#pragma unroll
  for (int n = 0; n < kN; ++n) { hstate[n] = 0.f; a[n] = -__expf(Alog[d * kN + n]); }
  const float dparam = Dp[d];
  for (int t = 0; t < kL; ++t) {
    size_t row = (size_t)(b * kL + t);
    if (tid < kN)            sB[tid]       = proj[row * kPP + kDTR + tid];
    else if (tid < 2 * kN)   sC[tid - kN]  = proj[row * kPP + kDTR + kN + (tid - kN)];
    __syncthreads();
    float dtv = dt[row * kDI + d];
    float u   = xc[row * kDI + d];
    float du  = dtv * u;
    float y = 0.f;
#pragma unroll
    for (int n = 0; n < kN; ++n) {
      float dA = __expf(dtv * a[n]);
      hstate[n] = dA * hstate[n] + du * sB[n];
      y += hstate[n] * sC[n];
    }
    float zv = xz[row * (size_t)(2 * kDI) + kDI + d];
    float g = zv / (1.f + __expf(-zv));
    y_bf[row * kDI + d] = f2bf((y + dparam * u) * g);
    __syncthreads();
  }
}

// =====================================================================
extern "C" void kernel_launch(void* const* d_in, const int* in_sizes, int n_in,
                              void* d_out, int out_size, void* d_ws, size_t ws_size,
                              hipStream_t stream)
{
  const float* x         = (const float*)d_in[0];
  const float* stat      = (const float*)d_in[1];
  /* d_in[2] time, d_in[4] labels: unused by reference output */
  const int*   mask      = (const int*)d_in[3];
  const float* emb_w     = (const float*)d_in[5];
  const float* emb_b     = (const float*)d_in[6];
  const float* stat_w    = (const float*)d_in[7];
  const float* stat_b    = (const float*)d_in[8];
  const float* ln_w      = (const float*)d_in[9];
  const float* ln_b      = (const float*)d_in[10];
  const float* norm_w    = (const float*)d_in[11];
  const float* in_proj_w = (const float*)d_in[12];
  const float* conv_w    = (const float*)d_in[13];
  const float* conv_b    = (const float*)d_in[14];
  const float* x_proj_w  = (const float*)d_in[15];
  const float* dt_proj_w = (const float*)d_in[16];
  const float* dt_proj_b = (const float*)d_in[17];
  const float* A_log     = (const float*)d_in[18];
  const float* D_param   = (const float*)d_in[19];
  const float* out_proj_w= (const float*)d_in[20];
  const float* norm_f_w  = (const float*)d_in[21];
  const float* lm_head_w = (const float*)d_in[22];
  float* out = (float*)d_out;

  char* p = (char*)d_ws;
  auto carve = [&](size_t bytes) -> char* {
    char* r = p; p += (bytes + 255) & ~(size_t)255; return r;
  };
  float*          h        = (float*)carve((size_t)kM * kD * 4);
  unsigned short* hn_bf    = (unsigned short*)carve((size_t)kM * kD * 2);
  float*          xz       = (float*)carve((size_t)kM * 2 * kDI * 4);
  float*          xc       = (float*)carve((size_t)kM * kDI * 4);
  unsigned short* xc_bf    = (unsigned short*)carve((size_t)kM * kDI * 2);
  float*          proj     = (float*)carve((size_t)kM * kPP * 4);
  unsigned short* dtr_bf   = (unsigned short*)carve((size_t)kM * kDTRP * 2);
  float*          dtb      = (float*)carve((size_t)kM * kDI * 4);
  unsigned short* y_bf     = (unsigned short*)carve((size_t)kM * kDI * 2);
  unsigned short* w_inproj = (unsigned short*)carve((size_t)kNL * 2 * kDI * kD * 2);
  unsigned short* w_xproj  = (unsigned short*)carve((size_t)kNL * kPP * kDI * 2);
  unsigned short* w_dtproj = (unsigned short*)carve((size_t)kNL * kDI * kDTRP * 2);
  unsigned short* w_outproj= (unsigned short*)carve((size_t)kNL * kD * kDI * 2);
  unsigned short* w_lmhead = (unsigned short*)carve((size_t)kV * kD * 2);
  (void)ws_size; (void)in_sizes; (void)n_in; (void)out_size;

  auto cvt = [&](const float* s, unsigned short* dst, size_t n) {
    f32_to_bf16<<<dim3((unsigned)((n + 255) / 256)), dim3(256), 0, stream>>>(s, dst, n);
  };
  cvt(in_proj_w,  w_inproj,  (size_t)kNL * 2 * kDI * kD);
  cvt(out_proj_w, w_outproj, (size_t)kNL * kD * kDI);
  cvt(lm_head_w,  w_lmhead,  (size_t)kV * kD);
  {  // dt_proj weights: (NL*DI) rows, columns padded 48 -> 64 with zeros
    size_t n = (size_t)kNL * kDI * kDTRP;
    f32_to_bf16_padcols<<<dim3((unsigned)((n + 255) / 256)), dim3(256), 0, stream>>>(
        dt_proj_w, w_dtproj, (size_t)kNL * kDI, kDTR, kDTRP, kDTR);
  }
  for (int l = 0; l < kNL; ++l) {  // x_proj weights: rows padded 80 -> 128 with zeros
    size_t n = (size_t)kPP * kDI;
    f32_to_bf16_padtile<<<dim3((unsigned)((n + 255) / 256)), dim3(256), 0, stream>>>(
        x_proj_w + (size_t)l * kP * kDI, w_xproj + (size_t)l * kPP * kDI,
        (size_t)kP, (size_t)kPP, kDI);
  }

  embed_layernorm<<<dim3(kM), dim3(256), 0, stream>>>(
      x, stat, mask, emb_w, emb_b, stat_w, stat_b, ln_w, ln_b, h);

  const size_t nTok = (size_t)kM * kDI;
  for (int l = 0; l < kNL; ++l) {
    rmsnorm_to_bf16<<<dim3(kM), dim3(256), 0, stream>>>(h, norm_w + (size_t)l * kD, hn_bf);
    gemm_bf16_wmma<0, 4, 2><<<dim3(2 * kDI / 64, kM / 32), dim3(32), 0, stream>>>(
        hn_bf, w_inproj + (size_t)l * 2 * kDI * kD, xz, kD, kD, kD, 2 * kDI);
    conv_silu<<<dim3((unsigned)(nTok / 256)), dim3(256), 0, stream>>>(
        xz, conv_w + (size_t)l * kDI * kConvK, conv_b + (size_t)l * kDI, xc, xc_bf);
    gemm_bf16_wmma<0, 4, 2><<<dim3(kPP / 64, kM / 32), dim3(32), 0, stream>>>(
        xc_bf, w_xproj + (size_t)l * kPP * kDI, proj, kDI, kDI, kDI, kPP);
    {  // dt_r = proj[..., :48] -> bf16, rows padded to stride 64 (zero tail)
      size_t n = (size_t)kM * kDTRP;
      f32_to_bf16_padcols<<<dim3((unsigned)((n + 255) / 256)), dim3(256), 0, stream>>>(
          proj, dtr_bf, (size_t)kM, kDTR, kDTRP, kPP);
    }
    gemm_bf16_wmma<0, 4, 2><<<dim3(kDI / 64, kM / 32), dim3(32), 0, stream>>>(
        dtr_bf, w_dtproj + (size_t)l * kDI * kDTRP, dtb, kDTRP, kDTRP, kDTRP, kDI);
    dt_softplus<<<dim3((unsigned)(nTok / 256)), dim3(256), 0, stream>>>(
        dtb, dt_proj_b + (size_t)l * kDI);
    selective_scan<<<dim3(kDI / 256, kB), dim3(256), 0, stream>>>(
        xc, dtb, xz, proj, A_log + (size_t)l * kDI * kN, D_param + (size_t)l * kDI, y_bf);
    gemm_bf16_wmma<1, 4, 2><<<dim3(kD / 64, kM / 32), dim3(32), 0, stream>>>(
        y_bf, w_outproj + (size_t)l * kD * kDI, h, kDI, kDI, kDI, kD);
  }
  rmsnorm_to_bf16<<<dim3(kM), dim3(256), 0, stream>>>(h, norm_f_w, hn_bf);
  gemm_bf16_wmma<0, 2, 2><<<dim3(kV / 32, kM / 32), dim3(32), 0, stream>>>(
      hn_bf, w_lmhead, out, kD, kD, kD, kV);
}
